// GRUTimeSeriesModel_9689446220036
// MI455X (gfx1250) — matrix operand order repaired and checked by reading, not compiled, not run on hardware
//
#include <hip/hip_runtime.h>
#include <cstddef>
#include <cstdint>

// ---------------------------------------------------------------------------
// GRU (4 layers, H=100, T=512, B=256, D=64) + dense head, for gfx1250.
// Per-layer persistent kernel: 16 WGs (16 batch rows each), 8 waves/WG
// splitting 19 gate N-tiles. bf16 WMMA 16x16x32, fp32 accumulate.
// x_t tiles streamed into LDS by the Tensor Data Mover (double-buffered,
// waited with s_wait_tensorcnt); weight B-fragments live in registers.
// ---------------------------------------------------------------------------

typedef __attribute__((ext_vector_type(16))) __bf16 v16bf;
typedef __attribute__((ext_vector_type(8)))  __bf16 v8bf;
typedef __attribute__((ext_vector_type(8)))  float  v8f;
typedef __attribute__((ext_vector_type(4)))  unsigned int v4u;
typedef __attribute__((ext_vector_type(8)))  int    v8i;
typedef __attribute__((ext_vector_type(4)))  int    v4i;

#define HID   100
#define G3    300
#define NPAD  304     // 19 tiles * 16
#define KPAD  128     // 4 K-tiles * 32
#define TLEN  512
#define BATCH 256
#define NT    19

// A-fragment (16x32 bf16) from LDS tile stored row-major [16][KPAD].
// Contiguous K order (not the ISA table order): lanes 0-15 hold M=0-15; elems e -> K = half*16 + e
// (matches B fragment).  Two aligned 16B LDS loads.
__device__ __forceinline__ v16bf load_a_frag(const __bf16* base, int kt, int lane) {
    int m = lane & 15;
    int half = lane >> 4;
    const v8bf* lo = (const v8bf*)(base + m * KPAD + kt * 32 + half * 16);
    const v8bf* hi = (const v8bf*)(base + m * KPAD + kt * 32 + half * 16 + 8);
    v8bf a = *lo, b = *hi;
    v16bf r;
#pragma unroll
    for (int i = 0; i < 8; ++i) { r[i] = a[i]; r[i + 8] = b[i]; }
    return r;
}

// B-fragment (32x16 bf16) from global padded weights [NPAD][KPAD] (row = N).
// Contiguous K order (same as A): lane's column N = lane&15; elems e -> K = 16*half + e.
// => one contiguous, 32B-aligned 32-byte load per lane.
__device__ __forceinline__ v16bf load_b_frag(const __bf16* w, int tile, int kt, int lane) {
    int n = tile * 16 + (lane & 15);
    int half = lane >> 4;
    return *(const v16bf*)(w + (size_t)n * KPAD + kt * 32 + half * 16);
}

// ---------------------------------------------------------------------------
// Weight prep: fp32 -> bf16, padded [layer][NPAD][KPAD], zeros outside.
// ---------------------------------------------------------------------------
__global__ void gru_prep_weights(const float* __restrict__ w_ih0,
                                 const float* __restrict__ w_ih_rest,
                                 const float* __restrict__ w_hh,
                                 __bf16* __restrict__ wih_pad,
                                 __bf16* __restrict__ whh_pad) {
    const int total = 4 * NPAD * KPAD;
    int idx = blockIdx.x * blockDim.x + threadIdx.x;
    if (idx >= 2 * total) return;
    int fam = idx / total;              // 0 = W_ih family, 1 = W_hh family
    int rem = idx - fam * total;
    int l = rem / (NPAD * KPAD);
    int r2 = rem - l * NPAD * KPAD;
    int n = r2 / KPAD;
    int k = r2 - n * KPAD;
    float v = 0.0f;
    if (fam == 0) {
        if (l == 0) { if (n < G3 && k < 64)  v = w_ih0[n * 64 + k]; }
        else        { if (n < G3 && k < HID) v = w_ih_rest[((size_t)(l - 1) * G3 + n) * HID + k]; }
        wih_pad[(size_t)l * NPAD * KPAD + n * KPAD + k] = (__bf16)v;
    } else {
        if (n < G3 && k < HID) v = w_hh[((size_t)l * G3 + n) * HID + k];
        whh_pad[(size_t)l * NPAD * KPAD + n * KPAD + k] = (__bf16)v;
    }
}

// Layer-0 input fp32 -> bf16 so every layer feeds the TDM the same way.
__global__ void gru_prep_x(const float* __restrict__ x, __bf16* __restrict__ xbf) {
    size_t i = (size_t)blockIdx.x * blockDim.x + threadIdx.x;
    if (i < (size_t)BATCH * TLEN * 64) xbf[i] = (__bf16)x[i];
}

// ---------------------------------------------------------------------------
// One GRU layer. grid = 16 WGs (16 batch rows each), block = 256 (8 waves).
// ---------------------------------------------------------------------------
__global__ __launch_bounds__(256) void gru_layer_kernel(
    const __bf16* __restrict__ xin,      // [B][T][D] bf16
    int D,                               // input feature dim (64 or 100)
    const __bf16* __restrict__ wih,      // [NPAD][KPAD] bf16 (zero padded)
    const __bf16* __restrict__ whh,      // [NPAD][KPAD] bf16 (zero padded)
    const float* __restrict__ b_ih,      // [300]
    const float* __restrict__ b_hh,      // [300]
    __bf16* __restrict__ out_act)        // [B][T][HID] bf16
{
    __shared__ float xg_s[16][NPAD];                       // x-gates (fp32)
    __shared__ float hg_s[16][NPAD];                       // h-gates (fp32)
    __shared__ __attribute__((aligned(32))) __bf16 x_a[2][16][KPAD]; // TDM dest
    __shared__ __attribute__((aligned(32))) __bf16 h_a[16][KPAD];    // A staging: h
    __shared__ float h_f[16][HID];                         // fp32 hidden state
    __shared__ float bI[G3], bH[G3];

    const int tid  = threadIdx.x;
    const int lane = tid & 31;
    const int swave = __builtin_amdgcn_readfirstlane(tid >> 5); // scalar wave id
    const int m0   = blockIdx.x * 16;    // batch base row

    // TDM: load 16xD bf16 tile (batch rows m0.., timestep t) into x_a[buf].
    // tile_dim0 = 128 with tensor_dim0 = D => cols D..127 read OOB -> zeros,
    // giving both K-padding and the 128-elem LDS row stride. 2-D tile.
    const uint64_t gbase = (uint64_t)(uintptr_t)xin + (uint64_t)m0 * TLEN * D * 2;
    const unsigned row_stride_elems = (unsigned)(TLEN * D);

    auto issue_tdm = [&](int t, int buf) {
        uint64_t ga = gbase + (uint64_t)t * D * 2;
        unsigned lds = (unsigned)(uintptr_t)&x_a[buf][0][0];
        v4u g0;
        g0[0] = 1u;                                  // count=1, load descriptor
        g0[1] = lds;                                 // lds_addr
        g0[2] = (unsigned)ga;                        // global_addr[31:0]
        g0[3] = (unsigned)(ga >> 32) | (2u << 30);   // global_addr[56:32] | type=2
        v8i g1;
        g1[0] = (int)(1u << 16);                     // wg_mask=0, data_size=2B
        g1[1] = (int)((unsigned)D << 16);            // tensor_dim0 = D (lo 16)
        g1[2] = (int)(16u << 16);                    // tensor_dim0 hi=0 | tensor_dim1=16
        g1[3] = (int)(128u << 16);                   // tensor_dim1 hi=0 | tile_dim0=128
        g1[4] = 16;                                  // tile_dim1=16, tile_dim2=0
        g1[5] = (int)row_stride_elems;               // tensor_dim0_stride (elems)
        g1[6] = 0;                                   // stride hi | dim1_stride lo
        g1[7] = 0;
        v4i z4 = {0, 0, 0, 0};
        v8i z8 = {0, 0, 0, 0, 0, 0, 0, 0};
        // 6-arg (clang-23 / therock) form: groups 2,3 + extra group zeroed (2-D tile)
        __builtin_amdgcn_tensor_load_to_lds(g0, g1, z4, z4, z8, 0);
    };

    // Kick off t=0 tile load, then init state while it flies.
    if (swave == 0) issue_tdm(0, 0);

    for (int u = tid; u < 16 * KPAD; u += 256) (&h_a[0][0])[u] = (__bf16)0.0f;
    for (int u = tid; u < 16 * HID; u += 256) (&h_f[0][0])[u] = 0.0f;
    for (int u = tid; u < G3; u += 256) { bI[u] = b_ih[u]; bH[u] = b_hh[u]; }

    // Loop-invariant weight fragments -> registers. Wave w owns tiles
    // {w, w+8, w+16} (tile < 19); guards are scalar (s_cbranch, EXEC all-1s).
    v16bf wBih[3][4], wBhh[3][4];
#pragma unroll
    for (int s = 0; s < 3; ++s) {
        int tile = swave + 8 * s;
        int tl = (tile < NT) ? tile : 0;
#pragma unroll
        for (int kt = 0; kt < 4; ++kt) {
            wBih[s][kt] = load_b_frag(wih, tl, kt, lane);
            wBhh[s][kt] = load_b_frag(whh, tl, kt, lane);
        }
    }
    if (swave == 0) __builtin_amdgcn_s_wait_tensorcnt(0);
    __syncthreads();   // x_a[0] ready & visible

    for (int t = 0; t < TLEN; ++t) {
        const int cur = t & 1, nxt = cur ^ 1;
        // Prefetch t+1 tile via TDM while we compute on t (disjoint buffer).
        if (swave == 0 && t + 1 < TLEN) issue_tdm(t + 1, nxt);

        // ---- WMMA: xg = x_t*W_ih^T, hg = h*W_hh^T (K padded to 128) ----
        v16bf aX[4], aH[4];
#pragma unroll
        for (int kt = 0; kt < 4; ++kt) {
            aX[kt] = load_a_frag(&x_a[cur][0][0], kt, lane);
            aH[kt] = load_a_frag(&h_a[0][0], kt, lane);
        }
#pragma unroll
        for (int s = 0; s < 3; ++s) {
            if (swave + 8 * s >= NT) continue;     // scalar guard
            v8f accX = {0.f, 0.f, 0.f, 0.f, 0.f, 0.f, 0.f, 0.f};
            v8f accH = {0.f, 0.f, 0.f, 0.f, 0.f, 0.f, 0.f, 0.f};
#pragma unroll
            for (int kt = 0; kt < 4; ++kt) {
                accX = __builtin_amdgcn_wmma_f32_16x16x32_bf16(
                    false, aX[kt], false, wBih[s][kt], (short)0, accX, false, false);
                accH = __builtin_amdgcn_wmma_f32_16x16x32_bf16(
                    false, aH[kt], false, wBhh[s][kt], (short)0, accH, false, false);
            }
            int n0 = (swave + 8 * s) * 16 + (lane & 15);
            int half = lane >> 4;
#pragma unroll
            for (int r = 0; r < 8; ++r) {          // D layout: M = r + 8*half
                xg_s[r + 8 * half][n0] = accX[r];
                hg_s[r + 8 * half][n0] = accH[r];
            }
        }
        __syncthreads();

        // ---- fused gate nonlinearity + state update (fp32) ----
        for (int u = tid; u < 16 * HID; u += 256) {
            int i = u / HID, j = u - i * HID;
            float xr = xg_s[i][j]       + bI[j];
            float hr = hg_s[i][j]       + bH[j];
            float xz = xg_s[i][HID + j] + bI[HID + j];
            float hz = hg_s[i][HID + j] + bH[HID + j];
            float xn = xg_s[i][2*HID+j] + bI[2*HID + j];
            float hn = hg_s[i][2*HID+j] + bH[2*HID + j];
            float rg = 1.0f / (1.0f + __expf(-(xr + hr)));
            float zg = 1.0f / (1.0f + __expf(-(xz + hz)));
            float ng = tanhf(xn + rg * hn);
            float hnew = (1.0f - zg) * ng + zg * h_f[i][j];
            h_f[i][j] = hnew;
            h_a[i][j] = (__bf16)hnew;
            out_act[((size_t)(m0 + i) * TLEN + t) * HID + j] = (__bf16)hnew;
        }
        // Ensure the prefetched x tile has landed before next iteration reads it.
        if (swave == 0 && t + 1 < TLEN) __builtin_amdgcn_s_wait_tensorcnt(0);
        __syncthreads();
    }
}

// ---------------------------------------------------------------------------
// Dense head: out[b, tt] = act[b, 416+tt, :] . dense_w + dense_b  (tiny)
// ---------------------------------------------------------------------------
__global__ void gru_dense_kernel(const __bf16* __restrict__ act,
                                 const float* __restrict__ dw,
                                 const float* __restrict__ db,
                                 float* __restrict__ out) {
    int idx = blockIdx.x * blockDim.x + threadIdx.x;
    if (idx >= BATCH * 96) return;
    int b = idx / 96, tt = idx - b * 96;
    const __bf16* row = act + ((size_t)b * TLEN + (TLEN - 96 + tt)) * HID;
    float s = db[0];
#pragma unroll 4
    for (int j = 0; j < HID; ++j) s += (float)row[j] * dw[j];
    out[idx] = s;
}

// ---------------------------------------------------------------------------
extern "C" void kernel_launch(void* const* d_in, const int* in_sizes, int n_in,
                              void* d_out, int out_size, void* d_ws, size_t ws_size,
                              hipStream_t stream) {
    const float* x         = (const float*)d_in[0]; // [256,512,64]
    const float* w_ih0     = (const float*)d_in[1]; // [300,64]
    const float* w_ih_rest = (const float*)d_in[2]; // [3,300,100]
    const float* w_hh      = (const float*)d_in[3]; // [4,300,100]
    const float* b_ih      = (const float*)d_in[4]; // [4,300]
    const float* b_hh      = (const float*)d_in[5]; // [4,300]
    const float* dense_w   = (const float*)d_in[6]; // [1,100]
    const float* dense_b   = (const float*)d_in[7]; // [1]
    float* out = (float*)d_out;                     // [256,96,1]

    // Workspace (bf16): padded weights + bf16 x + ping-pong activations (~70 MB)
    const size_t PADW = (size_t)4 * NPAD * KPAD;
    const size_t XSZ  = (size_t)BATCH * TLEN * 64;
    const size_t ACT  = (size_t)BATCH * TLEN * HID;
    __bf16* wih_pad = (__bf16*)d_ws;
    __bf16* whh_pad = wih_pad + PADW;
    __bf16* xbf     = whh_pad + PADW;
    __bf16* act0    = xbf + XSZ;
    __bf16* act1    = act0 + ACT;

    gru_prep_weights<<<(int)((2 * PADW + 255) / 256), 256, 0, stream>>>(
        w_ih0, w_ih_rest, w_hh, wih_pad, whh_pad);
    gru_prep_x<<<(int)((XSZ + 255) / 256), 256, 0, stream>>>(x, xbf);

    // Layer 0: D=64
    gru_layer_kernel<<<BATCH / 16, 256, 0, stream>>>(
        xbf, 64, wih_pad, whh_pad, b_ih, b_hh, act0);

    // Layers 1..3: D=100, ping-pong
    __bf16* bufs[2] = {act0, act1};
    for (int l = 1; l < 4; ++l) {
        gru_layer_kernel<<<BATCH / 16, 256, 0, stream>>>(
            bufs[(l + 1) & 1], HID,
            wih_pad + (size_t)l * NPAD * KPAD,
            whh_pad + (size_t)l * NPAD * KPAD,
            b_ih + l * G3, b_hh + l * G3, bufs[l & 1]);
    }

    gru_dense_kernel<<<(BATCH * 96 + 255) / 256, 256, 0, stream>>>(
        bufs[1], dense_w, dense_b, out);
}
